// PureHBNoActMLP_18098992185373
// MI455X (gfx1250) — compile-verified, matrix operand order checked by reading, not run on hardware
//
#include <hip/hip_runtime.h>

#define HID   1024
#define DIN   784
#define KPAD  800        // 25 * 32, zero-padded K
#define KSTEP 32
#define NITER 25
#define MTILE 32
#define CLS   10
#define LLAYERS 10
#define TOTALP 2046      // sum over layers of 2 * (512 >> l)
#define EPSV 1e-7f

typedef __attribute__((ext_vector_type(16))) __bf16 v16bf;
typedef __attribute__((ext_vector_type(8)))  float  v8f;

union FragU { v16bf v; __bf16 h[16]; uint4 q[2]; };

// Kernel 0: convert fc0_w (1024 x 784 f32) -> bf16, zero-padded to (1024 x 800)
__global__ __launch_bounds__(256) void wconv(const float* __restrict__ w,
                                             unsigned short* __restrict__ wbf) {
    int idx = blockIdx.x * 256 + threadIdx.x;
    if (idx >= HID * KPAD) return;
    int n = idx / KPAD;
    int k = idx - n * KPAD;
    float v = (k < DIN) ? w[n * DIN + k] : 0.0f;
    __bf16 b = (__bf16)v;                                  // native RNE convert
    wbf[idx] = __builtin_bit_cast(unsigned short, b);
}

// Fused: h = x@W^T + b  ->  hyper_butterfly(p1)  ->  hyper_butterfly(p2)  ->  @class_w + cb
__global__ __launch_bounds__(512) void fused(const float* __restrict__ x,
                                             const unsigned short* __restrict__ wbf,
                                             const float* __restrict__ bias,
                                             const float* __restrict__ p1,
                                             const float* __restrict__ p2,
                                             const float* __restrict__ cw,
                                             const float* __restrict__ cb,
                                             float* __restrict__ out) {
    __shared__ float hl[MTILE][HID];                       // 128 KB h tile
    __shared__ __align__(16) float albuf[2][MTILE][36];    // double-buffered x tile (stride 36: conflict-free)
    __shared__ float pl1[TOTALP];
    __shared__ float pl2[TOTALP];

    const int tid  = threadIdx.x;
    const int wave = tid >> 5;
    const int lane = tid & 31;
    const int m0   = blockIdx.x * MTILE;
    const int n0   = wave * 64;                 // wave's 64-column slice (4 N-subtiles)
    const int arow = lane & 15;
    const int ksel = lane >> 4;                 // which K half-chunk this lane holds

    // stage butterfly params into LDS (fenced by the first K-loop barrier)
    for (int i = tid; i < TOTALP; i += 512) { pl1[i] = p1[i]; pl2[i] = p2[i]; }

    v8f acc[2][4];
#pragma unroll
    for (int ms = 0; ms < 2; ++ms)
#pragma unroll
        for (int t = 0; t < 4; ++t) acc[ms][t] = (v8f)0.0f;

    // async staging of the 32x32 f32 x tile: 256 lanes x one b128 chunk each
    const int srow   = tid >> 3;                // 0..31 (for tid<256)
    const int schunk = tid & 7;                 // 0..7, 4 floats each
    const float* sga = x + (long)(m0 + srow) * DIN + schunk * 4;

#define ISSUE_A(kt)                                                                     \
    do {                                                                                \
        if (tid < 256 && (((kt) < NITER - 1) || (schunk < 4))) {                        \
            unsigned _lds = (unsigned)(size_t)&albuf[(kt) & 1][srow][schunk * 4];       \
            unsigned long long _ga = (unsigned long long)(const void*)(sga + (kt) * KSTEP); \
            asm volatile("global_load_async_to_lds_b128 %0, %1, off"                    \
                         :: "v"(_lds), "v"(_ga) : "memory");                            \
        }                                                                               \
    } while (0)

    ISSUE_A(0);

    // ---------------- Phase 1: GEMM via v_wmma_f32_16x16x32_bf16 ----------------
    for (int it = 0; it < NITER; ++it) {
        asm volatile("s_wait_asynccnt 0x0" ::: "memory");   // own chunks of buf[it&1] landed
        __syncthreads();                                    // everyone's chunks visible
        if (it + 1 < NITER) ISSUE_A(it + 1);                // prefetch next tile during compute

        // issue ALL B loads for this K-step first (latency overlapped by A conversion below)
        const int k0 = it * KSTEP;
        FragU bf[4];
#pragma unroll
        for (int t = 0; t < 4; ++t) {
            int n = n0 + t * 16 + arow;
            const unsigned short* bp = wbf + (long)n * KPAD + k0 + ksel * 8;
            bf[t].q[0] = *(const uint4*)bp;           // K = base .. base+7
            bf[t].q[1] = *(const uint4*)(bp + 16);    // K = base+16 .. base+23
        }

        // build two A fragments (M-subtiles) from LDS f32, native packed bf16 converts
        FragU af[2];
#pragma unroll
        for (int ms = 0; ms < 2; ++ms) {
            const float* base = &albuf[it & 1][ms * 16 + arow][ksel * 8];
#pragma unroll
            for (int j = 0; j < 8; ++j) af[ms].h[j]     = (__bf16)base[j];        // K = base..+7
#pragma unroll
            for (int j = 0; j < 8; ++j) af[ms].h[8 + j] = (__bf16)base[16 + j];   // K = base+16..+23
            if (it == NITER - 1) af[ms].q[1] = make_uint4(0u, 0u, 0u, 0u);        // k >= 784 -> 0
        }

#pragma unroll
        for (int t = 0; t < 4; ++t)
#pragma unroll
            for (int ms = 0; ms < 2; ++ms)        // each B fragment feeds 2 WMMAs
                acc[ms][t] = __builtin_amdgcn_wmma_f32_16x16x32_bf16(
                    false, af[ms].v, false, bf[t].v, (short)0, acc[ms][t], false, false);
    }

    // scatter accumulators (+bias) into the LDS h tile
#pragma unroll
    for (int ms = 0; ms < 2; ++ms)
#pragma unroll
        for (int t = 0; t < 4; ++t) {
            int n = n0 + t * 16 + arow;
            float bv = bias[n];
#pragma unroll
            for (int r = 0; r < 8; ++r)
                hl[ms * 16 + ksel * 8 + r][n] = acc[ms][t][r] + bv;
        }
    __syncthreads();

    // ---------------- Phase 2: per-row hyperbolic butterfly (wave-local) ----------------
    const float sc = 0.03162277660168379f;        // sqrt(0.001)

    for (int rr = 0; rr < 2; ++rr) {
        const int row = wave * 2 + rr;
        float* u = hl[row];

        for (int blk = 0; blk < 2; ++blk) {
            const float* pp = (blk == 0) ? pl1 : pl2;

            // log_map_zero
            float ss = 0.0f;
            for (int j = lane; j < HID; j += 32) { float v = u[j]; ss += v * v; }
#pragma unroll
            for (int o = 16; o; o >>= 1) ss += __shfl_xor(ss, o, 32);
            float nn = fmaxf(sqrtf(ss), EPSV);
            float z  = fminf(fmaxf(sc * nn, EPSV), 1.0f - 1e-6f);
            float s  = 0.5f * __logf((1.0f + z) / (1.0f - z)) / (sc * nn);
            for (int j = lane; j < HID; j += 32) u[j] *= s;

            // 10 butterfly layers; 512 disjoint pairs/layer, 16 per lane
            int off = 0;
#pragma unroll
            for (int l = 0; l < LLAYERS; ++l) {
                const int stride = 1 << l;
                for (int p = lane; p < 512; p += 32) {
                    int bI = p >> l;
                    int w  = p & (stride - 1);
                    int i1 = (bI << (l + 1)) + w;
                    int i2 = i1 + stride;
                    float aP = pp[off + 2 * bI];
                    float bP = pp[off + 2 * bI + 1];
                    float x1 = u[i1], x2 = u[i2];
                    u[i1] =  aP * x1 + bP * x2;
                    u[i2] = -bP * x1 + aP * x2;
                }
                off += 2 * (512 >> l);
            }

            // exp_map_zero
            ss = 0.0f;
            for (int j = lane; j < HID; j += 32) { float v = u[j]; ss += v * v; }
#pragma unroll
            for (int o = 16; o; o >>= 1) ss += __shfl_xor(ss, o, 32);
            nn = fmaxf(sqrtf(ss), EPSV);
            s  = tanhf(sc * nn) / (sc * nn);
            for (int j = lane; j < HID; j += 32) u[j] *= s;
        }

        // classifier: 1024 x 10, shuffle-reduced per wave
        float part[CLS];
#pragma unroll
        for (int c = 0; c < CLS; ++c) part[c] = 0.0f;
        for (int j = lane; j < HID; j += 32) {
            float v = u[j];
#pragma unroll
            for (int c = 0; c < CLS; ++c) part[c] += v * cw[j * CLS + c];
        }
#pragma unroll
        for (int c = 0; c < CLS; ++c) {
            float s2 = part[c];
#pragma unroll
            for (int o = 16; o; o >>= 1) s2 += __shfl_xor(s2, o, 32);
            if (lane == 0) out[(long)(m0 + row) * CLS + c] = s2 + cb[c];
        }
    }
}

extern "C" void kernel_launch(void* const* d_in, const int* in_sizes, int n_in,
                              void* d_out, int out_size, void* d_ws, size_t ws_size,
                              hipStream_t stream) {
    const float* x   = (const float*)d_in[0];
    const float* w   = (const float*)d_in[1];
    const float* b   = (const float*)d_in[2];
    const float* p1  = (const float*)d_in[3];
    const float* p2  = (const float*)d_in[4];
    const float* cw  = (const float*)d_in[5];
    const float* cb  = (const float*)d_in[6];
    float* out = (float*)d_out;

    unsigned short* wbf = (unsigned short*)d_ws;   // 1024*800*2 = 1.6 MB scratch

    const int nconv = HID * KPAD;
    wconv<<<(nconv + 255) / 256, 256, 0, stream>>>(w, wbf);

    const int batch = in_sizes[0] / DIN;           // 65536
    fused<<<batch / MTILE, 512, 0, stream>>>(x, wbf, b, p1, p2, cw, cb, out);
}
